// OwnSasrecModel_7352984010962
// MI455X (gfx1250) — matrix operand order repaired and checked by reading, not compiled, not run on hardware
//
#include <hip/hip_runtime.h>
#include <hip/hip_bf16.h>

#define EPS       1e-3f
#define NEGV      (-4294967295.0f)   // -2^32+1 rounded in f32
#define NITEMS    200000
#define BSZ       512
#define SSZ       64
#define DSZ       64
#define NROWS     (BSZ*SSZ)          // 32768

typedef __attribute__((ext_vector_type(2)))  float        v2f;
typedef __attribute__((ext_vector_type(8)))  float        v8f;
typedef __attribute__((ext_vector_type(16))) __bf16       v16bf;
typedef __attribute__((ext_vector_type(4)))  unsigned int u32x4;
struct U32x8 { u32x4 lo, hi; };

static __device__ __forceinline__ float wave_sum(float x) {
    #pragma unroll
    for (int m = 16; m > 0; m >>= 1) x += __shfl_xor(x, m, 32);
    return x;
}
static __device__ __forceinline__ float wave_max(float x) {
    #pragma unroll
    for (int m = 16; m > 0; m >>= 1) x = fmaxf(x, __shfl_xor(x, m, 32));
    return x;
}

// ---------------------------------------------------------------------------
// seq[b,s,:] = (item_emb[id] + pos_emb[s]) * mask ; also store row mask
// ---------------------------------------------------------------------------
__global__ void sasrec_embed(const int* __restrict__ ids,
                             const float* __restrict__ item_emb,
                             const float* __restrict__ pos_emb,
                             float* __restrict__ seq,
                             float* __restrict__ rowmask) {
    int row = blockIdx.x;            // 0..32767
    int d   = threadIdx.x;           // 0..63
    int s   = row & (SSZ - 1);
    int id  = ids[row];
    float m = (id != NITEMS) ? 1.f : 0.f;
    seq[(size_t)row * DSZ + d] = (item_emb[(size_t)id * DSZ + d] + pos_emb[s * DSZ + d]) * m;
    if (d == 0) rowmask[row] = m;
}

// ---------------------------------------------------------------------------
// Row LayerNorm over 64 features, one wave per row (wave32: 2 elems/lane)
// ---------------------------------------------------------------------------
__global__ __launch_bounds__(128) void sasrec_ln(const float* __restrict__ in,
                                                 const float* __restrict__ g,
                                                 const float* __restrict__ bb,
                                                 float* __restrict__ out,
                                                 int nrows, int in_stride) {
    int wave = threadIdx.x >> 5;
    int lane = threadIdx.x & 31;
    int row  = blockIdx.x * 4 + wave;
    if (row >= nrows) return;
    const float* p = in + (size_t)row * in_stride;
    float x0 = p[lane], x1 = p[lane + 32];
    float s  = wave_sum(x0 + x1);
    float sq = wave_sum(x0 * x0 + x1 * x1);
    float mu  = s * (1.f / 64.f);
    float var = sq * (1.f / 64.f) - mu * mu;
    float rs  = rsqrtf(var + EPS);
    float* o = out + (size_t)row * DSZ;
    o[lane]      = (x0 - mu) * rs * g[lane]      + bb[lane];
    o[lane + 32] = (x1 - mu) * rs * g[lane + 32] + bb[lane + 32];
}

// ---------------------------------------------------------------------------
// C[M x 64] = epilogue(A[M x 64] @ W[64 x 64] + bias [+R] [relu] [*rowmask])
// One 16x16 C tile per wave via V_WMMA_F32_16X16X4_F32, K-loop of 16.
// ---------------------------------------------------------------------------
__global__ __launch_bounds__(256) void sasrec_gemm64(const float* __restrict__ A,
                                                     const float* __restrict__ W,
                                                     const float* __restrict__ bias,
                                                     const float* __restrict__ R,
                                                     const float* __restrict__ rowmask,
                                                     float* __restrict__ C,
                                                     int M, int relu) {
    __shared__ float sW[64 * 65];
    __shared__ float sB[64];
    int tid = threadIdx.x;
    for (int i = tid; i < 64 * 64; i += 256) sW[(i >> 6) * 65 + (i & 63)] = W[i];
    if (tid < 64) sB[tid] = bias[tid];
    __syncthreads();

    int wave = tid >> 5, lane = tid & 31;
    int t  = blockIdx.x * 8 + wave;
    int m0 = (t >> 2) << 4;
    int n0 = (t & 3) << 4;
    if (m0 >= M) return;                     // wave-uniform: EXEC stays all-1s

    int laneHi = lane >> 4;                  // 0 / 1
    int lm     = lane & 15;
    int kb     = laneHi * 2;
    const float* arow = A + (size_t)(m0 + lm) * 64;

    v8f acc = {0.f, 0.f, 0.f, 0.f, 0.f, 0.f, 0.f, 0.f};
    #pragma unroll
    for (int k = 0; k < 64; k += 4) {
        v2f a = *(const v2f*)(arow + k + kb);                 // A[M][k+kb], A[M][k+kb+1]
        v2f b;
        b.x = sW[(k + kb)     * 65 + n0 + lm];                // W[k+kb][N]
        b.y = sW[(k + kb + 1) * 65 + n0 + lm];                // W[k+kb+1][N]
        acc = __builtin_amdgcn_wmma_f32_16x16x4_f32(false, a, false, b,
                                                    (short)0, acc, false, false);
    }

    int mo = m0 + laneHi * 8;
    int nc = n0 + lm;
    #pragma unroll
    for (int r = 0; r < 8; ++r) {
        float v = acc[r] + sB[nc];
        if (R)       v += R[(size_t)(mo + r) * 64 + nc];
        if (relu)    v  = fmaxf(v, 0.f);
        if (rowmask) v *= rowmask[mo + r];
        C[(size_t)(mo + r) * 64 + nc] = v;
    }
}

// ---------------------------------------------------------------------------
// Per-batch attention: scores=QK^T/8, key+causal mask, softmax, query mask,
// O = P@V + X1(residual).  One workgroup per batch, 8 query rows per wave.
// ---------------------------------------------------------------------------
__global__ __launch_bounds__(256) void sasrec_attn(const float* __restrict__ Q,
                                                   const float* __restrict__ K,
                                                   const float* __restrict__ V,
                                                   const float* __restrict__ X1,
                                                   const int* __restrict__ ids,
                                                   float* __restrict__ XO) {
    __shared__ float sQ[64 * 65], sK[64 * 65], sV[64 * 65];
    __shared__ float sM[64];
    __shared__ float sP[8 * 64];
    int b = blockIdx.x, tid = threadIdx.x;
    size_t base = (size_t)b * (SSZ * DSZ);
    for (int i = tid; i < SSZ * DSZ; i += 256) {
        int r = i >> 6, c = i & 63;
        sQ[r * 65 + c] = Q[base + i];
        sK[r * 65 + c] = K[base + i];
        sV[r * 65 + c] = V[base + i];
    }
    if (tid < 64) sM[tid] = (ids[b * 64 + tid] != NITEMS) ? 1.f : 0.f;
    __syncthreads();

    int wave = tid >> 5, lane = tid & 31;
    float* pp = sP + wave * 64;
    for (int qi = wave * 8; qi < wave * 8 + 8; ++qi) {
        const float* qrow = sQ + qi * 65;
        float s0 = 0.f, s1 = 0.f;
        #pragma unroll 8
        for (int d = 0; d < 64; ++d) {
            float qv = qrow[d];
            s0 += qv * sK[lane * 65 + d];
            s1 += qv * sK[(lane + 32) * 65 + d];
        }
        s0 *= 0.125f; s1 *= 0.125f;
        int k0 = lane, k1 = lane + 32;
        if (k0 > qi || sM[k0] == 0.f) s0 = NEGV;
        if (k1 > qi || sM[k1] == 0.f) s1 = NEGV;
        float mx    = wave_max(fmaxf(s0, s1));
        float e0    = __expf(s0 - mx), e1 = __expf(s1 - mx);
        float inv   = 1.f / wave_sum(e0 + e1);
        pp[k0] = e0 * inv;
        pp[k1] = e1 * inv;            // same-wave LDS RAW: DS ops are in-order
        float o0 = 0.f, o1 = 0.f;
        #pragma unroll 8
        for (int k = 0; k < 64; ++k) {
            float p = pp[k];          // broadcast read
            o0 += p * sV[k * 65 + lane];
            o1 += p * sV[k * 65 + lane + 32];
        }
        float qm = sM[qi];
        size_t ob = base + (size_t)qi * 64;
        XO[ob + lane]      = o0 * qm + X1[ob + lane];
        XO[ob + lane + 32] = o1 * qm + X1[ob + lane + 32];
    }
}

// ---------------------------------------------------------------------------
// f32 -> bf16 (round to nearest even); source is streamed once -> NT load
// ---------------------------------------------------------------------------
__global__ void sasrec_cvt_bf16(const float* __restrict__ in,
                                unsigned short* __restrict__ out, int n) {
    int i = blockIdx.x * 256 + threadIdx.x;
    if (i >= n) return;
    float f = __builtin_nontemporal_load(in + i);
    unsigned int u = __float_as_uint(f);
    u += 0x7FFFu + ((u >> 16) & 1u);
    out[i] = (unsigned short)(u >> 16);   // bf16 table is reused -> cached store
}

// ---------------------------------------------------------------------------
// out[512 x 200000] = Aemb[512x64] @ items[200000x64]^T  (bf16 WMMA, f32 acc)
// One wave: one 16-row M tile x four 16-col N tiles; 2 wmma per tile (K=64).
// C is written once and never read: nontemporal stores keep the 25.6MB bf16
// item table resident in L2 instead of being evicted by 410MB of output.
// ---------------------------------------------------------------------------
static __device__ __forceinline__ v16bf load_fragA(const __bf16* rowptr, int koff) {
    U32x8 u;                                   // halves [koff,koff+8) and [koff+16,koff+24)
    u.lo = *(const u32x4*)(rowptr + koff);
    u.hi = *(const u32x4*)(rowptr + koff + 16);
    return __builtin_bit_cast(v16bf, u);
}
static __device__ __forceinline__ v16bf load_fragB(const __bf16* rowptr, int koff) {
    U32x8 u;                                   // halves [koff,koff+8) and [koff+16,koff+24)
    u.lo = *(const u32x4*)(rowptr + koff);
    u.hi = *(const u32x4*)(rowptr + koff + 16);
    return __builtin_bit_cast(v16bf, u);
}

__global__ __launch_bounds__(256) void sasrec_score(const __bf16* __restrict__ Aemb,
                                                    const __bf16* __restrict__ items,
                                                    float* __restrict__ out) {
    int tid = threadIdx.x, wave = tid >> 5, lane = tid & 31;
    int t  = blockIdx.x * 8 + wave;            // 0..99999 = 32 Mtiles * 3125 Ngroups
    int mt = t / 3125;
    int ng = t - mt * 3125;
    int m0 = mt << 4;
    int nbase = ng << 6;
    int lm = lane & 15, laneHi = lane >> 4;

    const __bf16* arow = Aemb + (size_t)(m0 + lm) * 64;
    int ka  = laneHi * 8;                      // A: interleave-by-8 K pattern
    v16bf a0 = load_fragA(arow, ka);           // K 0..31
    v16bf a1 = load_fragA(arow, ka + 32);      // K 32..63
    int kbB = laneHi * 8;                      // B: interleave-by-8 K pattern

    #pragma unroll
    for (int j = 0; j < 4; ++j) {
        int n0 = nbase + (j << 4);
        const __bf16* brow = items + (size_t)(n0 + lm) * 64;
        v16bf b0 = load_fragB(brow, kbB);
        v16bf b1 = load_fragB(brow, kbB + 32);
        v8f acc = {0.f, 0.f, 0.f, 0.f, 0.f, 0.f, 0.f, 0.f};
        acc = __builtin_amdgcn_wmma_f32_16x16x32_bf16(false, a0, false, b0,
                                                      (short)0, acc, false, false);
        acc = __builtin_amdgcn_wmma_f32_16x16x32_bf16(false, a1, false, b1,
                                                      (short)0, acc, false, false);
        int mo = m0 + laneHi * 8;
        #pragma unroll
        for (int r = 0; r < 8; ++r)
            __builtin_nontemporal_store(acc[r],
                out + (size_t)(mo + r) * NITEMS + n0 + lm);
    }
}

// ---------------------------------------------------------------------------
extern "C" void kernel_launch(void* const* d_in, const int* in_sizes, int n_in,
                              void* d_out, int out_size, void* d_ws, size_t ws_size,
                              hipStream_t stream) {
    (void)in_sizes; (void)n_in; (void)out_size; (void)ws_size;
    const int*   ids      = (const int*)d_in[0];
    const float* item_emb = (const float*)d_in[1];
    const float* pos_emb  = (const float*)d_in[2];
    const float* ln1_g = (const float*)d_in[3];
    const float* ln1_b = (const float*)d_in[4];
    const float* Wq = (const float*)d_in[5];
    const float* bq = (const float*)d_in[6];
    const float* Wk = (const float*)d_in[7];
    const float* bk = (const float*)d_in[8];
    const float* Wv = (const float*)d_in[9];
    const float* bv = (const float*)d_in[10];
    const float* ln2_g = (const float*)d_in[11];
    const float* ln2_b = (const float*)d_in[12];
    const float* W1 = (const float*)d_in[13];
    const float* b1 = (const float*)d_in[14];
    const float* W2 = (const float*)d_in[15];
    const float* b2 = (const float*)d_in[16];
    const float* lnf_g = (const float*)d_in[17];
    const float* lnf_b = (const float*)d_in[18];
    const float* out_items = (const float*)d_in[19];
    float* out = (float*)d_out;

    char* ws = (char*)d_ws;
    const size_t SZ = (size_t)NROWS * DSZ * sizeof(float);     // 8 MB per activation
    float* seq = (float*)(ws + 0 * SZ);
    float* x1  = (float*)(ws + 1 * SZ);
    float* q   = (float*)(ws + 2 * SZ);
    float* kk  = (float*)(ws + 3 * SZ);
    float* v   = (float*)(ws + 4 * SZ);
    float* xo  = (float*)(ws + 5 * SZ);
    float* x2  = (float*)(ws + 6 * SZ);
    float* h   = q;                                            // q free after attn
    float*          rowmask  = (float*)(ws + 7 * SZ);
    float*          semb     = (float*)(ws + 7 * SZ + 0x40000);
    unsigned short* semb_bf  = (unsigned short*)(ws + 7 * SZ + 0x80000);
    unsigned short* items_bf = (unsigned short*)(ws + 7 * SZ + 0xC0000);

    sasrec_embed<<<NROWS, 64, 0, stream>>>(ids, item_emb, pos_emb, seq, rowmask);
    for (int i = 0; i < 2; ++i) {
        sasrec_ln<<<NROWS / 4, 128, 0, stream>>>(seq, ln1_g + i * 64, ln1_b + i * 64, x1, NROWS, 64);
        sasrec_gemm64<<<NROWS / 32, 256, 0, stream>>>(x1,  Wq + i * 4096, bq + i * 64, nullptr, nullptr, q,  NROWS, 0);
        sasrec_gemm64<<<NROWS / 32, 256, 0, stream>>>(seq, Wk + i * 4096, bk + i * 64, nullptr, nullptr, kk, NROWS, 0);
        sasrec_gemm64<<<NROWS / 32, 256, 0, stream>>>(seq, Wv + i * 4096, bv + i * 64, nullptr, nullptr, v,  NROWS, 0);
        sasrec_attn<<<BSZ, 256, 0, stream>>>(q, kk, v, x1, ids, xo);
        sasrec_ln<<<NROWS / 4, 128, 0, stream>>>(xo, ln2_g + i * 64, ln2_b + i * 64, x2, NROWS, 64);
        sasrec_gemm64<<<NROWS / 32, 256, 0, stream>>>(x2, W1 + i * 4096, b1 + i * 64, nullptr, nullptr, h, NROWS, 1);
        sasrec_gemm64<<<NROWS / 32, 256, 0, stream>>>(h,  W2 + i * 4096, b2 + i * 64, x2, rowmask, seq, NROWS, 0);
    }
    // final LN on last position only (rows stride S*D), then bf16 convert + score
    sasrec_ln<<<BSZ / 4, 128, 0, stream>>>(seq + 63 * 64, lnf_g, lnf_b, semb, BSZ, SSZ * DSZ);
    sasrec_cvt_bf16<<<(BSZ * DSZ + 255) / 256, 256, 0, stream>>>(semb, semb_bf, BSZ * DSZ);
    sasrec_cvt_bf16<<<(NITEMS * DSZ + 255) / 256, 256, 0, stream>>>(out_items, items_bf, NITEMS * DSZ);
    sasrec_score<<<12500, 256, 0, stream>>>((const __bf16*)semb_bf, (const __bf16*)items_bf, out);
}
